// ShardedExperts_8469675508184
// MI455X (gfx1250) — compile-verified, matrix operand order checked by reading, not run on hardware
//
#include <hip/hip_runtime.h>
#include <hip/hip_bf16.h>

// ---------------------------------------------------------------------------
// MoE sharded-experts partial forward for MI455X (gfx1250, wave32).
//   out = sum_{e in {0,2,4,6}} w_e ⊙ ( silu(x@Wg_e^T) * (x@Wu_e^T) ) @ Wd_e^T
// Compute-bound (~824 GFLOP dense vs ~14us HBM traffic at 23.3 TB/s), so:
//  - pass 0 converts fp32 -> bf16 once (X) / per expert (Wgu, Wd) into ws
//  - GEMMs run v_wmma_f32_16x16x32_bf16 with double-buffered LDS tiles staged
//    by global_load_async_to_lds_b128 (ASYNCcnt pipeline, no VGPR round-trip)
//  - K-loop software pipeline is peeled: steady state has no branches around
//    the stage/wait, last slice waits asynccnt==0.
// ---------------------------------------------------------------------------

#define T_TOK   8192
#define D_HID   2048
#define I_INT   2048
#define N_LOCAL 4        // local expert ids: 0,2,4,6

#define BM 128
#define BN 64
#define BK 64
#define NTHREADS 256     // 8 wave32s: 4 (M) x 2 (N) wave grid, 32x32 per wave

typedef __attribute__((ext_vector_type(16))) __bf16 v16bf;
typedef __attribute__((ext_vector_type(8)))  __bf16 v8bf;
typedef __attribute__((ext_vector_type(8)))  float  v8f;

// ---------------- CDNA5 async global->LDS helpers --------------------------
// Encoding (CDNA5 ISA / CDNA5 ISA): VDST = per-lane LDS
// byte address VGPR, VADDR = 64-bit global address. The low 32 bits of a
// generic pointer to __shared__ memory are the LDS byte address (flat
// aperture lives in addr[63:32]).
__device__ inline void async_copy_b128(void* lds_ptr, const void* gptr) {
  unsigned int lds_addr = (unsigned int)(unsigned long long)lds_ptr;
  asm volatile("global_load_async_to_lds_b128 %0, %1, off"
               :: "v"(lds_addr), "v"(gptr) : "memory");
}

#if __has_builtin(__builtin_amdgcn_s_wait_asynccnt)
#define WAIT_ASYNC(n) __builtin_amdgcn_s_wait_asynccnt(n)
#else
#define WAIT_ASYNC(n) asm volatile("s_wait_asynccnt %0" :: "i"(n) : "memory")
#endif

// Load a 16x32 bf16 WMMA operand fragment from an LDS tile stored row-major
// with row stride BK. Per ISA layout (CDNA5 ISA §7.12.2):
//   lanes 0-15 : row = row0+lane,    K-chunks {kk..kk+7} and {kk+16..kk+23}
//   lanes 16-31: row = row0+lane-16, K-chunks {kk+8..kk+15} and {kk+24..kk+31}
__device__ inline v16bf ldsFrag(const __bf16* base, int row0, int kk, int lane) {
  const int half = lane >> 4;
  const int r    = row0 + (lane & 15);
  const __bf16* p = base + r * BK + kk + half * 8;    // 16B aligned
  const v8bf lo = *reinterpret_cast<const v8bf*>(p);
  const v8bf hi = *reinterpret_cast<const v8bf*>(p + 16);
  v16bf f;
#pragma unroll
  for (int i = 0; i < 8; ++i) { f[i] = lo[i]; f[i + 8] = hi[i]; }
  return f;
}

__device__ inline v8f wmma_bf16(v16bf a, v16bf b, v8f c) {
  // (neg_a, A, neg_b, B, c_mod, C, reuse_a, reuse_b)
  return __builtin_amdgcn_wmma_f32_16x16x32_bf16(false, a, false, b,
                                                 (short)0, c, false, false);
}

__device__ inline float fast_sigmoid(float x) {
  return 1.0f / (1.0f + __builtin_amdgcn_exp2f(-1.442695041f * x));
}

// ---------------------------------------------------------------------------
// Kernel: fp32 -> bf16 conversion (8 elements per thread-step).
// ---------------------------------------------------------------------------
__global__ __launch_bounds__(256)
void f32_to_bf16(const float* __restrict__ src, __bf16* __restrict__ dst, int n8) {
  int i = blockIdx.x * 256 + threadIdx.x;
  const int stride = gridDim.x * 256;
  for (; i < n8; i += stride) {
    const float4 a = *reinterpret_cast<const float4*>(src + (size_t)i * 8);
    const float4 b = *reinterpret_cast<const float4*>(src + (size_t)i * 8 + 4);
    v8bf o;
    o[0] = (__bf16)a.x; o[1] = (__bf16)a.y; o[2] = (__bf16)a.z; o[3] = (__bf16)a.w;
    o[4] = (__bf16)b.x; o[5] = (__bf16)b.y; o[6] = (__bf16)b.z; o[7] = (__bf16)b.w;
    *reinterpret_cast<v8bf*>(dst + (size_t)i * 8) = o;
  }
}

// ---------------------------------------------------------------------------
// Kernel: per-token routing weight for each local expert.
// ---------------------------------------------------------------------------
__global__ __launch_bounds__(256)
void moe_gate_weights(const int* __restrict__ sel, const float* __restrict__ rw,
                      float* __restrict__ wout) {
  const int t = blockIdx.x * 256 + threadIdx.x;
  if (t >= T_TOK) return;
  const int   s0 = sel[2 * t], s1 = sel[2 * t + 1];
  const float r0 = rw[2 * t],  r1 = rw[2 * t + 1];
#pragma unroll
  for (int i = 0; i < N_LOCAL; ++i) {
    const int eid = 2 * i;                       // LOCAL_IDS = (0,2,4,6)
    wout[i * T_TOK + t] = (s0 == eid ? r0 : 0.f) + (s1 == eid ? r1 : 0.f);
  }
}

// ---------------------------------------------------------------------------
// Kernel: zero the output accumulator (d_out is poisoned by the harness).
// ---------------------------------------------------------------------------
__global__ __launch_bounds__(256)
void moe_zero_out(float4* __restrict__ p, int n4) {
  const float4 z = {0.f, 0.f, 0.f, 0.f};
  int i = blockIdx.x * blockDim.x + threadIdx.x;
  const int stride = gridDim.x * blockDim.x;
  for (; i < n4; i += stride) p[i] = z;
}

// ---------------------------------------------------------------------------
// Kernel (per expert): GEMM1 + SwiGLU.  H = silu(X@Wg^T) * (X@Wu^T), bf16.
// All operands pre-converted bf16; LDS tiles double-buffered and filled with
// async global->LDS b128 copies (8 async instructions per thread per stage).
// ---------------------------------------------------------------------------
__global__ __launch_bounds__(NTHREADS)
void moe_gemm1_swiglu(const __bf16* __restrict__ X, const __bf16* __restrict__ Wgu,
                      __bf16* __restrict__ H) {
  __shared__ __bf16 sA [2][BM * BK];   // 2 x 16 KB
  __shared__ __bf16 sBg[2][BN * BK];   // 2 x  8 KB
  __shared__ __bf16 sBu[2][BN * BK];   // 2 x  8 KB

  const int tid  = threadIdx.x;
  const int lane = tid & 31;
  const int wave = tid >> 5;
  const int wm0  = (wave >> 1) * 32;
  const int wn0  = (wave & 1) * 32;
  const int m_block = blockIdx.y * BM;
  const int n_block = blockIdx.x * BN;

  // Stage one BK-slice into LDS buffer b: pure bf16 b128 async copies.
  auto stage = [&](int b, int kt) {
#pragma unroll
    for (int it = 0; it < 4; ++it) {             // A: 128x64 bf16 = 1024 x 16B
      const int v   = tid + it * NTHREADS;
      const int row = v >> 3;
      const int cv  = (v & 7) << 3;
      async_copy_b128(&sA[b][row * BK + cv],
                      X + (size_t)(m_block + row) * D_HID + kt + cv);
    }
#pragma unroll
    for (int it = 0; it < 2; ++it) {             // Bg, Bu: 64x64 each = 512 x 16B
      const int v   = tid + it * NTHREADS;
      const int row = v >> 3;
      const int cv  = (v & 7) << 3;
      async_copy_b128(&sBg[b][row * BK + cv],
                      Wgu + (size_t)(n_block + row) * D_HID + kt + cv);
      async_copy_b128(&sBu[b][row * BK + cv],
                      Wgu + (size_t)(I_INT + n_block + row) * D_HID + kt + cv);
    }
  };

  const v8f zf = {0.f, 0.f, 0.f, 0.f, 0.f, 0.f, 0.f, 0.f};
  v8f accg[2][2], accu[2][2];
#pragma unroll
  for (int i = 0; i < 2; ++i)
#pragma unroll
    for (int j = 0; j < 2; ++j) { accg[i][j] = zf; accu[i][j] = zf; }

  auto compute = [&](int b) {
    const __bf16* cA  = sA [b];
    const __bf16* cBg = sBg[b];
    const __bf16* cBu = sBu[b];
#pragma unroll
    for (int kk = 0; kk < BK; kk += 32) {
      const v16bf a0 = ldsFrag(cA,  wm0,      kk, lane);
      const v16bf a1 = ldsFrag(cA,  wm0 + 16, kk, lane);
      const v16bf g0 = ldsFrag(cBg, wn0,      kk, lane);
      const v16bf g1 = ldsFrag(cBg, wn0 + 16, kk, lane);
      const v16bf u0 = ldsFrag(cBu, wn0,      kk, lane);
      const v16bf u1 = ldsFrag(cBu, wn0 + 16, kk, lane);
      accg[0][0] = wmma_bf16(a0, g0, accg[0][0]);
      accg[0][1] = wmma_bf16(a0, g1, accg[0][1]);
      accg[1][0] = wmma_bf16(a1, g0, accg[1][0]);
      accg[1][1] = wmma_bf16(a1, g1, accg[1][1]);
      accu[0][0] = wmma_bf16(a0, u0, accu[0][0]);
      accu[0][1] = wmma_bf16(a0, u1, accu[0][1]);
      accu[1][0] = wmma_bf16(a1, u0, accu[1][0]);
      accu[1][1] = wmma_bf16(a1, u1, accu[1][1]);
    }
  };

  // Software pipeline: steady state is branch-free (stage -> wait -> compute).
  stage(0, 0);
  int buf = 0;
  for (int kt = 0; kt < D_HID - BK; kt += BK) {
    stage(buf ^ 1, kt + BK);
    WAIT_ASYNC(8);          // 8 newer ops outstanding => current tile landed
    __syncthreads();
    compute(buf);
    __syncthreads();        // all waves done reading buf before it is restaged
    buf ^= 1;
  }
  WAIT_ASYNC(0);            // final slice
  __syncthreads();
  compute(buf);

  // Epilogue: h = silu(gate) * up -> bf16 workspace.
  // C/D layout: VGPR r, lanes 0-15 -> (M=r, N=lane); lanes 16-31 -> (M=8+r).
#pragma unroll
  for (int am = 0; am < 2; ++am)
#pragma unroll
    for (int bn = 0; bn < 2; ++bn) {
      const int ncol  = n_block + wn0 + bn * 16 + (lane & 15);
      const int mbase = m_block + wm0 + am * 16 + ((lane >> 4) << 3);
      const v8f g = accg[am][bn];
      const v8f u = accu[am][bn];
#pragma unroll
      for (int r = 0; r < 8; ++r) {
        const float gv = g[r];
        const float hv = gv * fast_sigmoid(gv) * u[r];
        H[(size_t)(mbase + r) * I_INT + ncol] = (__bf16)hv;
      }
    }
}

// ---------------------------------------------------------------------------
// Kernel (per expert): GEMM2 + gated accumulate.
//   Out[t, d] += w[t] * sum_i H[t, i] * Wd[d, i]
// Experts serialize on the stream and each block owns a distinct tile, so the
// plain read-modify-write is race-free.
// ---------------------------------------------------------------------------
__global__ __launch_bounds__(NTHREADS)
void moe_gemm2_down(const __bf16* __restrict__ H, const __bf16* __restrict__ Wd,
                    const float* __restrict__ Wt, float* __restrict__ Out) {
  __shared__ __bf16 sA[2][BM * BK];    // 2 x 16 KB
  __shared__ __bf16 sB[2][BN * BK];    // 2 x  8 KB

  const int tid  = threadIdx.x;
  const int lane = tid & 31;
  const int wave = tid >> 5;
  const int wm0  = (wave >> 1) * 32;
  const int wn0  = (wave & 1) * 32;
  const int m_block = blockIdx.y * BM;
  const int n_block = blockIdx.x * BN;   // over D_HID

  auto stage = [&](int b, int kt) {
#pragma unroll
    for (int it = 0; it < 4; ++it) {             // A: 128x64 bf16
      const int v   = tid + it * NTHREADS;
      const int row = v >> 3;
      const int cv  = (v & 7) << 3;
      async_copy_b128(&sA[b][row * BK + cv],
                      H + (size_t)(m_block + row) * I_INT + kt + cv);
    }
#pragma unroll
    for (int it = 0; it < 2; ++it) {             // B: 64x64 bf16
      const int v   = tid + it * NTHREADS;
      const int row = v >> 3;
      const int cv  = (v & 7) << 3;
      async_copy_b128(&sB[b][row * BK + cv],
                      Wd + (size_t)(n_block + row) * I_INT + kt + cv);
    }
  };

  const v8f zf = {0.f, 0.f, 0.f, 0.f, 0.f, 0.f, 0.f, 0.f};
  v8f acc[2][2];
#pragma unroll
  for (int i = 0; i < 2; ++i)
#pragma unroll
    for (int j = 0; j < 2; ++j) acc[i][j] = zf;

  auto compute = [&](int b) {
    const __bf16* cA = sA[b];
    const __bf16* cB = sB[b];
#pragma unroll
    for (int kk = 0; kk < BK; kk += 32) {
      const v16bf a0 = ldsFrag(cA, wm0,      kk, lane);
      const v16bf a1 = ldsFrag(cA, wm0 + 16, kk, lane);
      const v16bf b0 = ldsFrag(cB, wn0,      kk, lane);
      const v16bf b1 = ldsFrag(cB, wn0 + 16, kk, lane);
      acc[0][0] = wmma_bf16(a0, b0, acc[0][0]);
      acc[0][1] = wmma_bf16(a0, b1, acc[0][1]);
      acc[1][0] = wmma_bf16(a1, b0, acc[1][0]);
      acc[1][1] = wmma_bf16(a1, b1, acc[1][1]);
    }
  };

  stage(0, 0);
  int buf = 0;
  for (int kt = 0; kt < I_INT - BK; kt += BK) {
    stage(buf ^ 1, kt + BK);
    WAIT_ASYNC(6);
    __syncthreads();
    compute(buf);
    __syncthreads();
    buf ^= 1;
  }
  WAIT_ASYNC(0);
  __syncthreads();
  compute(buf);

  // Epilogue: Out += w[t] * acc
#pragma unroll
  for (int am = 0; am < 2; ++am)
#pragma unroll
    for (int bn = 0; bn < 2; ++bn) {
      const int ncol  = n_block + wn0 + bn * 16 + (lane & 15);
      const int mbase = m_block + wm0 + am * 16 + ((lane >> 4) << 3);
      const v8f a = acc[am][bn];
#pragma unroll
      for (int r = 0; r < 8; ++r) {
        const int m = mbase + r;
        const size_t o = (size_t)m * D_HID + ncol;
        Out[o] += Wt[m] * a[r];
      }
    }
}

// ---------------------------------------------------------------------------
// Host entry point. Everything on `stream` (graph-capture safe). Workspace:
//   [ 0MB, 32MB)  bf16 Xbf [T, D]
//   [32MB, 64MB)  bf16 Hbf [T, I]          (reused per expert)
//   [64MB, 80MB)  bf16 Wgu_bf [2I, D]      (reused per expert)
//   [80MB, 88MB)  bf16 Wd_bf  [D, I]       (reused per expert)
//   [88MB, +128K) float w[4][T]
// ---------------------------------------------------------------------------
extern "C" void kernel_launch(void* const* d_in, const int* in_sizes, int n_in,
                              void* d_out, int out_size, void* d_ws, size_t ws_size,
                              hipStream_t stream) {
  (void)in_sizes; (void)n_in; (void)out_size; (void)ws_size;

  const float* X   = (const float*)d_in[0];
  const int*   sel = (const int*)d_in[1];
  const float* rw  = (const float*)d_in[2];
  const float* Wgu = (const float*)d_in[3];
  const float* Wd  = (const float*)d_in[4];
  float*       Out = (float*)d_out;

  char* ws = (char*)d_ws;
  __bf16* Xbf   = (__bf16*)(ws);
  __bf16* Hbf   = (__bf16*)(ws + (size_t)T_TOK * D_HID * 2);
  __bf16* Wgubf = (__bf16*)(ws + (size_t)T_TOK * D_HID * 2 * 2);
  __bf16* Wdbf  = (__bf16*)(ws + (size_t)T_TOK * D_HID * 4 + (size_t)2 * I_INT * D_HID * 2);
  float*  Wwt   = (float*) (ws + (size_t)T_TOK * D_HID * 4 + (size_t)3 * I_INT * D_HID * 2);

  moe_zero_out<<<2048, 256, 0, stream>>>((float4*)d_out, T_TOK * D_HID / 4);
  moe_gate_weights<<<T_TOK / 256, 256, 0, stream>>>(sel, rw, Wwt);
  f32_to_bf16<<<4096, 256, 0, stream>>>(X, Xbf, T_TOK * D_HID / 8);

  const dim3 blk(NTHREADS);
  const dim3 g1(I_INT / BN, T_TOK / BM);   // 32 x 64
  const dim3 g2(D_HID / BN, T_TOK / BM);   // 32 x 64

  for (int e = 0; e < N_LOCAL; ++e) {
    const float* Wgu_e = Wgu + (size_t)e * 2 * I_INT * D_HID;
    const float* Wd_e  = Wd  + (size_t)e * D_HID * I_INT;
    f32_to_bf16<<<4096, 256, 0, stream>>>(Wgu_e, Wgubf, 2 * I_INT * D_HID / 8);
    f32_to_bf16<<<2048, 256, 0, stream>>>(Wd_e,  Wdbf,  D_HID * I_INT / 8);
    moe_gemm1_swiglu<<<g1, blk, 0, stream>>>(Xbf, Wgubf, Hbf);
    moe_gemm2_down <<<g2, blk, 0, stream>>>(Hbf, Wdbf, Wwt + (size_t)e * T_TOK, Out);
  }
}
